// SelectiveStateSpace_49400713838743
// MI455X (gfx1250) — compile-verified, matrix operand order checked by reading, not run on hardware
//
#include <hip/hip_runtime.h>
#include <math.h>

typedef __attribute__((ext_vector_type(16))) __bf16        v16bf;
typedef __attribute__((ext_vector_type(8)))  float         v8f;
typedef __attribute__((ext_vector_type(8)))  unsigned short v8u16;
typedef int vint4 __attribute__((vector_size(16)));

constexpr int BATCH = 16;
constexpr int SEQ   = 2048;
constexpr int HID   = 1024;

// ---------------------------------------------------------------- helpers ---

static __device__ __forceinline__ unsigned short bf16_of(float f) {
    unsigned u = __float_as_uint(f);
    u += 0x7FFFu + ((u >> 16) & 1u);          // round-to-nearest-even
    return (unsigned short)(u >> 16);
}

// ---- async global->LDS copy (gfx1250 GLOBAL_LOAD_ASYNC_TO_LDS_B128) --------
// Guarded: falls back to VGPR-staged copy if the builtin is absent.
// Probe round 2: builtin signature wants (int4 global*, int4 local*, imm, imm).

#if __has_builtin(__builtin_amdgcn_global_load_async_to_lds_b128)
#define HAVE_ASYNC_CP 1
#endif

static __device__ __forceinline__ void async_cp16(const void* g, void* l) {
#ifdef HAVE_ASYNC_CP
    __builtin_amdgcn_global_load_async_to_lds_b128(
        (__attribute__((address_space(1))) vint4*)g,
        (__attribute__((address_space(3))) vint4*)l, 0, 0);
#else
    *(v8u16*)l = *(const v8u16*)g;
#endif
}

template <int N>
static __device__ __forceinline__ void wait_async() {
#ifdef HAVE_ASYNC_CP
#if __has_builtin(__builtin_amdgcn_s_wait_asynccnt)
    __builtin_amdgcn_s_wait_asynccnt(N);
#else
    asm volatile("s_wait_asynccnt %0" ::"i"(N) : "memory");
#endif
#endif
}

// Load one 16x32 bf16 WMMA fragment (A or B operand; both are K-major here).
// ISA layout (CDNA5 ISA): lanes 0-15 = rows, K 0..7 & 16..23; lanes 16-31 =
// same rows, K 8..15 & 24..31.  Two ds/global b128 loads per lane.
static __device__ __forceinline__ v16bf load_frag(const unsigned short* base, int stride) {
    const int lane = threadIdx.x & 31;
    const unsigned short* p = base + (lane & 15) * stride + (lane >> 4) * 8;
    union { v8u16 h[2]; v16bf v; } u;
    u.h[0] = *(const v8u16*)(p);
    u.h[1] = *(const v8u16*)(p + 16);
    return u.v;
}

// ------------------------------------------------------------ prep kernel ---

__global__ void prep_kernel(const float* __restrict__ A, const float* __restrict__ Win,
                            const float* __restrict__ Wout,
                            const float* __restrict__ b_in, const float* __restrict__ sbias,
                            unsigned short* __restrict__ Abf, unsigned short* __restrict__ Winbf,
                            unsigned short* __restrict__ Woutbf,
                            float* __restrict__ biasIn, unsigned int* __restrict__ counter)
{
    int i = blockIdx.x * blockDim.x + threadIdx.x;
    if (i == 0) *counter = 0u;
    if (i < HID * HID) {
        Abf[i]    = bf16_of(A[i]);
        Winbf[i]  = bf16_of(Win[i]);
        Woutbf[i] = bf16_of(Wout[i]);
    }
    if (i < HID) biasIn[i] = b_in[i] + sbias[i];
}

// ----------------------------------------------------- big bookend GEMMs ----
// C[M,N] = A[M,K] * B[N,K]^T + bias  (both operands K-major -> "TN" GEMM).
// Double-buffered LDS tiles; bf16 tiles staged with async-to-LDS copies
// (ASYNCcnt), f32 A tiles converted through VGPRs.

constexpr int BM = 128, BN = 64, BK = 32, LDT = 40;   // LDT: padded LDS stride

template <bool A_IS_F32>
__global__ __launch_bounds__(256) void wmma_gemm_kernel(
    const void* __restrict__ Ain, const unsigned short* __restrict__ Bw,
    const float* __restrict__ bias, float* __restrict__ Cout,
    int M, int N, int K)
{
    __shared__ unsigned short sA[2][BM * LDT];   // 2 x 10 KB
    __shared__ unsigned short sB[2][BN * LDT];   // 2 x  5 KB

    const int tid  = threadIdx.x;
    const int lane = tid & 31;
    const int w    = tid >> 5;                // 8 waves
    const int wm   = w & 3, wn = w >> 2;      // 4x2 wave grid, 32x32 per wave
    const int m0   = blockIdx.y * BM;
    const int n0   = blockIdx.x * BN;

    // per-thread staging coordinates (chunks of 8 bf16 = 16B)
    const int ar0 = (tid +   0) >> 2, ac0 = ((tid +   0) & 3) * 8;
    const int ar1 = (tid + 256) >> 2, ac1 = ((tid + 256) & 3) * 8;
    const int br  = tid >> 2,         bc  = (tid & 3) * 8;

    auto stage = [&](int k0, int buf) {
        if (A_IS_F32) {
            const float* g0 = (const float*)Ain + (size_t)(m0 + ar0) * K + (k0 + ac0);
            const float* g1 = (const float*)Ain + (size_t)(m0 + ar1) * K + (k0 + ac1);
            v8u16 p0, p1;
            #pragma unroll
            for (int e = 0; e < 8; ++e) { p0[e] = bf16_of(g0[e]); p1[e] = bf16_of(g1[e]); }
            *(v8u16*)&sA[buf][ar0 * LDT + ac0] = p0;
            *(v8u16*)&sA[buf][ar1 * LDT + ac1] = p1;
        } else {
            async_cp16((const unsigned short*)Ain + (size_t)(m0 + ar0) * K + (k0 + ac0),
                       &sA[buf][ar0 * LDT + ac0]);
            async_cp16((const unsigned short*)Ain + (size_t)(m0 + ar1) * K + (k0 + ac1),
                       &sA[buf][ar1 * LDT + ac1]);
        }
        async_cp16(Bw + (size_t)(n0 + br) * K + (k0 + bc), &sB[buf][br * LDT + bc]);
    };

    // async ops in flight per stage; async loads retire in order, so waiting
    // "<= NASY" guarantees the PREVIOUS tile's copies have landed while the
    // just-issued prefetch stays in flight.
    constexpr int NASY = A_IS_F32 ? 1 : 3;
    const int KT = K / BK;

    v8f acc00 = {}, acc01 = {}, acc10 = {}, acc11 = {};

    stage(0, 0);
    for (int kt = 0; kt < KT; ++kt) {
        const int cur = kt & 1;
        if (kt + 1 < KT) { stage((kt + 1) * BK, cur ^ 1); wait_async<NASY>(); }
        else             { wait_async<0>(); }
        __syncthreads();

        v16bf a0 = load_frag(&sA[cur][(wm * 32 +  0) * LDT], LDT);
        v16bf a1 = load_frag(&sA[cur][(wm * 32 + 16) * LDT], LDT);
        v16bf b0 = load_frag(&sB[cur][(wn * 32 +  0) * LDT], LDT);
        v16bf b1 = load_frag(&sB[cur][(wn * 32 + 16) * LDT], LDT);
        acc00 = __builtin_amdgcn_wmma_f32_16x16x32_bf16(false, a0, false, b0, (short)0, acc00, false, false);
        acc01 = __builtin_amdgcn_wmma_f32_16x16x32_bf16(false, a0, false, b1, (short)0, acc01, false, false);
        acc10 = __builtin_amdgcn_wmma_f32_16x16x32_bf16(false, a1, false, b0, (short)0, acc10, false, false);
        acc11 = __builtin_amdgcn_wmma_f32_16x16x32_bf16(false, a1, false, b1, (short)0, acc11, false, false);
        __syncthreads();      // all reads of buf[cur] done before it is re-staged
    }

    // ---- epilogue: C layout lane = N, VGPR = M row (+8 for upper lanes) ----
    const int cn = lane & 15;
    const int mh = (lane >> 4) * 8;
    v8f accs[2][2] = {{acc00, acc01}, {acc10, acc11}};
    #pragma unroll
    for (int i = 0; i < 2; ++i)
        #pragma unroll
        for (int j = 0; j < 2; ++j) {
            int gm = m0 + wm * 32 + i * 16 + mh;
            int gn = n0 + wn * 32 + j * 16 + cn;
            float bv = bias[gn];
            float* o = Cout + (size_t)gm * N + gn;
            #pragma unroll
            for (int r = 0; r < 8; ++r) o[(size_t)r * N] = accs[i][j][r] + bv;
        }
}

// ------------------------------------------------------------ scan kernel ---
// Persistent: 16 WGs x 4 waves. Each wave owns a 16-wide N slice. h (16x1024)
// staged in LDS per step (async copies); recurrent matrix A streamed bf16 from
// L2. Grid-wide step barrier = monotonic atomic counter + s_sleep spin.

constexpr int SBLK = 16;          // workgroups
constexpr int SPH  = HID + 8;     // padded LDS stride (row = 2064 B, 16B-aligned)

__global__ __launch_bounds__(128) void scan_kernel(
    const unsigned short* __restrict__ Abf,
    const float* __restrict__ u,
    unsigned short* __restrict__ sOut,
    unsigned short* __restrict__ hbuf,        // 2 x (BATCH*HID) bf16, double buffer
    float* __restrict__ hFinal,
    unsigned int* __restrict__ counter)
{
    __shared__ unsigned short sH[BATCH * SPH];    // ~33 KB

    const int tid  = threadIdx.x;
    const int w    = tid >> 5;
    const int lane = tid & 31;
    const int n0   = blockIdx.x * (HID / SBLK) + w * 16;
    const unsigned short* Arow = Abf + (size_t)n0 * HID;   // B-operand rows n0..n0+15

    for (int t = 0; t < SEQ; ++t) {
        const unsigned short* hprev = hbuf + ((t    ) & 1) * (BATCH * HID);
        unsigned short*       hnext = hbuf + ((t + 1) & 1) * (BATCH * HID);

        // stage h_{t-1} into LDS (zeros at t==0)
        if (t == 0) {
            v8u16 z = {0, 0, 0, 0, 0, 0, 0, 0};
            for (int c = tid; c < BATCH * HID / 8; c += 128) {
                int row = c >> 7, col = (c & 127) * 8;
                *(v8u16*)&sH[row * SPH + col] = z;
            }
        } else {
            for (int c = tid; c < BATCH * HID / 8; c += 128) {
                int row = c >> 7, col = (c & 127) * 8;
                async_cp16(&hprev[row * HID + col], &sH[row * SPH + col]);
            }
            wait_async<0>();
        }
        __syncthreads();

        // K=1024 reduction: 32 WMMAs as 4 independent 8-deep chains
        v8f a0 = {}, a1 = {}, a2 = {}, a3 = {};
        #pragma unroll
        for (int kb = 0; kb < HID / 32; kb += 4) {
            v16bf h0 = load_frag(&sH[(kb + 0) * 32], SPH);
            v16bf w0 = load_frag(Arow + (kb + 0) * 32, HID);
            a0 = __builtin_amdgcn_wmma_f32_16x16x32_bf16(false, h0, false, w0, (short)0, a0, false, false);
            v16bf h1 = load_frag(&sH[(kb + 1) * 32], SPH);
            v16bf w1 = load_frag(Arow + (kb + 1) * 32, HID);
            a1 = __builtin_amdgcn_wmma_f32_16x16x32_bf16(false, h1, false, w1, (short)0, a1, false, false);
            v16bf h2 = load_frag(&sH[(kb + 2) * 32], SPH);
            v16bf w2 = load_frag(Arow + (kb + 2) * 32, HID);
            a2 = __builtin_amdgcn_wmma_f32_16x16x32_bf16(false, h2, false, w2, (short)0, a2, false, false);
            v16bf h3 = load_frag(&sH[(kb + 3) * 32], SPH);
            v16bf w3 = load_frag(Arow + (kb + 3) * 32, HID);
            a3 = __builtin_amdgcn_wmma_f32_16x16x32_bf16(false, h3, false, w3, (short)0, a3, false, false);
        }
        v8f acc = (a0 + a1) + (a2 + a3);

        // epilogue: + u_t, exact GELU, requantize state to bf16
        const int gn = n0 + (lane & 15);
        const int mh = (lane >> 4) * 8;
        #pragma unroll
        for (int r = 0; r < 8; ++r) {
            int b = mh + r;
            float v = acc[r] + u[((size_t)b * SEQ + t) * HID + gn];
            float g = 0.5f * v * (1.0f + erff(v * 0.70710678118654752f));
            unsigned short gb = bf16_of(g);
            hnext[b * HID + gn]                     = gb;
            sOut[((size_t)b * SEQ + t) * HID + gn]  = gb;
            if (t == SEQ - 1) hFinal[b * HID + gn]  = g;
        }
        __syncthreads();

        // grid-wide step barrier (release/acquire, monotonic counter)
        __threadfence();
        if (tid == 0) {
            __hip_atomic_fetch_add(counter, 1u, __ATOMIC_ACQ_REL, __HIP_MEMORY_SCOPE_AGENT);
            const unsigned target = (unsigned)(t + 1) * (unsigned)gridDim.x;
            while (__hip_atomic_load(counter, __ATOMIC_ACQUIRE, __HIP_MEMORY_SCOPE_AGENT) < target)
                __builtin_amdgcn_s_sleep(2);
        }
        __syncthreads();
        __threadfence();
    }
}

// --------------------------------------------------------------- launcher ---

extern "C" void kernel_launch(void* const* d_in, const int* in_sizes, int n_in,
                              void* d_out, int out_size, void* d_ws, size_t ws_size,
                              hipStream_t stream)
{
    (void)in_sizes; (void)n_in; (void)out_size; (void)ws_size;

    const float* x     = (const float*)d_in[0];
    const float* A     = (const float*)d_in[1];
    const float* Win   = (const float*)d_in[2];
    const float* b_in  = (const float*)d_in[3];
    const float* sbias = (const float*)d_in[4];
    const float* Wout  = (const float*)d_in[5];
    const float* b_out = (const float*)d_in[6];

    char* ws = (char*)d_ws;
    unsigned int* counter = (unsigned int*)ws;
    size_t off = 256;
    float*          u      = (float*)(ws + off);          off += (size_t)BATCH * SEQ * HID * 4;
    unsigned short* s      = (unsigned short*)(ws + off); off += (size_t)BATCH * SEQ * HID * 2;
    unsigned short* hbuf   = (unsigned short*)(ws + off); off += (size_t)2 * BATCH * HID * 2;
    unsigned short* Abf    = (unsigned short*)(ws + off); off += (size_t)HID * HID * 2;
    unsigned short* Winbf  = (unsigned short*)(ws + off); off += (size_t)HID * HID * 2;
    unsigned short* Woutbf = (unsigned short*)(ws + off); off += (size_t)HID * HID * 2;
    float*          biasIn = (float*)(ws + off);          off += (size_t)HID * 4;

    float* out    = (float*)d_out;
    float* hFinal = out + (size_t)BATCH * SEQ * HID;

    // 1) weight conversion + bias fuse + barrier reset
    prep_kernel<<<(HID * HID + 255) / 256, 256, 0, stream>>>(
        A, Win, Wout, b_in, sbias, Abf, Winbf, Woutbf, biasIn, counter);

    // 2) u = x @ W_in^T + (b_in + state_bias)      [M=32768, N=K=1024]
    dim3 g1(HID / BN, (BATCH * SEQ) / BM);
    wmma_gemm_kernel<true><<<g1, 256, 0, stream>>>(
        x, Winbf, biasIn, u, BATCH * SEQ, HID, HID);

    // 3) sequential scan (persistent, grid barrier per step)
    scan_kernel<<<SBLK, 128, 0, stream>>>(Abf, u, s, hbuf, hFinal, counter);

    // 4) out = h @ W_out^T + b_out
    wmma_gemm_kernel<false><<<g1, 256, 0, stream>>>(
        s, Woutbf, b_out, out, BATCH * SEQ, HID, HID);
}
